// ScaledDotProductAttention_14328010899869
// MI455X (gfx1250) — matrix operand order repaired and checked by reading, not compiled, not run on hardware
//
#include <hip/hip_runtime.h>

typedef _Float16 h8   __attribute__((ext_vector_type(8)));
typedef _Float16 v16h __attribute__((ext_vector_type(16)));
typedef float    v8f  __attribute__((ext_vector_type(8)));
typedef float    f8   __attribute__((ext_vector_type(8)));

// Q pre-scale: log2(e)/sqrt(50) -> scores land directly in log2 domain,
// so softmax needs only exp2(s - m), no per-element log2e multiply.
#define QSCALE_LOG2 0.20403990900592843f   // 1.4426950408889634 / sqrt(50)

__device__ __forceinline__ v16h mk16(h8 lo, h8 hi) {
  v16h r;
#pragma unroll
  for (int i = 0; i < 8; ++i) { r[i] = lo[i]; r[i + 8] = hi[i]; }
  return r;
}

__device__ __forceinline__ v8f vzero8() {
  v8f z;
#pragma unroll
  for (int i = 0; i < 8; ++i) z[i] = 0.0f;
  return z;
}

// ---------------------------------------------------------------------------
// Phase 1: QKV projection via WMMA.
//   grid = (32768/128, 3), block = 256 (8 waves, 16 tokens each)
//   blockIdx.y selects q/k/v weight slice.
//   qh/kh: [32768][64] f16 (head dim zero-padded 50->64; q pre-scaled by
//   log2e/sqrt(50)); v stored transposed: [8][64][4096] f16.
// ---------------------------------------------------------------------------
#define KT_STRIDE 264   // 256 + 8 halves pad: conflict-free & 16B-aligned rows

__global__ __launch_bounds__(256) void qkv_kernel(
    const float* __restrict__ x, const float* __restrict__ wmat,
    _Float16* __restrict__ qh, _Float16* __restrict__ kh,
    _Float16* __restrict__ vt) {
  __shared__ _Float16 kt[64 * KT_STRIDE];

  const int wsel = blockIdx.y;                 // 0=q 1=k 2=v
  const float* wk = wmat + (size_t)wsel * 256 * 50;

  // Stage weight slice into LDS, transposed to [h][d], f16, zero pad h>=50.
  for (int idx = threadIdx.x; idx < 64 * 256; idx += 256) {
    int d = idx & 255, h = idx >> 8;
    float val = (h < 50) ? wk[d * 50 + h] : 0.0f;
    kt[h * KT_STRIDE + d] = (_Float16)val;
  }
  __syncthreads();

  const int wave = threadIdx.x >> 5;
  const int lane = threadIdx.x & 31;
  const int ln = lane & 15, hl = lane >> 4;
  const int tokbase = blockIdx.x * 128 + wave * 16;

  // A operands: x rows, f32 -> f16. Interleaved A K-pattern:
  // lane<16: K {0..7}U{16..23}; lane>=16: K {8..15}U{24..31} (+32*s).
  const float* xr = x + (size_t)(tokbase + ln) * 256;
  v16h a[8];
#pragma unroll
  for (int s = 0; s < 8; ++s) {
    h8 lo = __builtin_convertvector(*(const f8*)(xr + s * 32 + hl * 8), h8);
    h8 hi = __builtin_convertvector(*(const f8*)(xr + s * 32 + hl * 8 + 16), h8);
    a[s] = mk16(lo, hi);
  }

  const float scale = (wsel == 0) ? QSCALE_LOG2 : 1.0f;
#pragma unroll
  for (int nt = 0; nt < 4; ++nt) {
    const int h = nt * 16 + ln;                 // B column = head dim
    const _Float16* krow = &kt[h * KT_STRIDE];
    v8f acc = vzero8();
#pragma unroll
    for (int s = 0; s < 8; ++s) {
      // B interleaved K pattern: lane half hl holds K {hl*8..hl*8+7}U{16+hl*8..16+hl*8+7}.
      const _Float16* bp = krow + s * 32 + hl * 8;
      v16h bop = mk16(*(const h8*)bp, *(const h8*)(bp + 16));
      acc = __builtin_amdgcn_wmma_f32_16x16x32_f16(
          false, a[s], false, bop, (short)0, acc, false, false);
    }
    if (wsel == 2) {
#pragma unroll
      for (int r = 0; r < 8; ++r) {
        int tok = tokbase + r + (hl << 3);
        int bb = tok >> 12, tt = tok & 4095;
        vt[(size_t)bb * 64 * 4096 + (size_t)h * 4096 + tt] = (_Float16)acc[r];
      }
    } else {
      _Float16* dst = (wsel == 0) ? qh : kh;
#pragma unroll
      for (int r = 0; r < 8; ++r) {
        int tok = tokbase + r + (hl << 3);
        dst[(size_t)tok * 64 + h] = (_Float16)(acc[r] * scale);
      }
    }
  }
}

// ---------------------------------------------------------------------------
// Phase 2: flash attention. grid = (4096/128, 8 batches), block = 256.
// Each wave owns 16 query rows, iterates keys in blocks of 64.
// ---------------------------------------------------------------------------
#define P_STRIDE 72   // halves; 144B rows: 16B aligned, conflict-free

__global__ __launch_bounds__(256) void attn_kernel(
    const _Float16* __restrict__ qh, const _Float16* __restrict__ kh,
    const _Float16* __restrict__ vt, float* __restrict__ out) {
  __shared__ _Float16 pbuf[8 * 16 * P_STRIDE];

  const int wave = threadIdx.x >> 5;
  const int lane = threadIdx.x & 31;
  const int ln = lane & 15, hl = lane >> 4;
  const int b = blockIdx.y;
  const int q0 = blockIdx.x * 128 + wave * 16;

  // Q A-operands (interleaved A layout), head dim split 0..31 / 32..63.
  const _Float16* qrow = qh + (size_t)(b * 4096 + q0 + ln) * 64;
  v16h aq0 = mk16(*(const h8*)(qrow + hl * 8), *(const h8*)(qrow + hl * 8 + 16));
  v16h aq1 = mk16(*(const h8*)(qrow + 32 + hl * 8), *(const h8*)(qrow + 48 + hl * 8));

  const _Float16* kbb = kh + (size_t)b * 4096 * 64;
  const _Float16* vtb = vt + (size_t)b * 64 * 4096;
  _Float16* mylds = pbuf + wave * 16 * P_STRIDE;

  float m[8], l[8];
  v8f o[4];
#pragma unroll
  for (int r = 0; r < 8; ++r) { m[r] = -1.0e30f; l[r] = 0.0f; }
#pragma unroll
  for (int j = 0; j < 4; ++j) o[j] = vzero8();

  for (int kb = 0; kb < 64; ++kb) {
    const int key0 = kb * 64;

    // ---- scores (log2 domain): 4 key tiles of 16, K-dim = head 64 ----
    v8f s[4];
#pragma unroll
    for (int t = 0; t < 4; ++t) {
      const _Float16* krp = kbb + (size_t)(key0 + t * 16 + ln) * 64;
      v16h bk0 = mk16(*(const h8*)(krp + hl * 8), *(const h8*)(krp + hl * 8 + 16));
      v16h bk1 = mk16(*(const h8*)(krp + 32 + hl * 8), *(const h8*)(krp + 48 + hl * 8));
      v8f acc = vzero8();
      acc = __builtin_amdgcn_wmma_f32_16x16x32_f16(false, aq0, false, bk0,
                                                   (short)0, acc, false, false);
      acc = __builtin_amdgcn_wmma_f32_16x16x32_f16(false, aq1, false, bk1,
                                                   (short)0, acc, false, false);
      s[t] = acc;
    }

    // Prefetch next iteration's K rows / V segments into cache.
    if (kb < 63) {
      const _Float16* nk = kbb + (size_t)(key0 + 64) * 64;
      __builtin_prefetch(nk + (size_t)lane * 64, 0, 3);
      __builtin_prefetch(nk + (size_t)(lane + 32) * 64, 0, 3);
      __builtin_prefetch(vtb + (size_t)lane * 4096 + key0 + 64, 0, 3);
      __builtin_prefetch(vtb + (size_t)(lane + 32) * 4096 + key0 + 64, 0, 3);
    }

    // ---- online softmax (row = VGPR idx + 8*half; N across 16 lanes) ----
    float mx[8];
#pragma unroll
    for (int r = 0; r < 8; ++r)
      mx[r] = fmaxf(fmaxf(s[0][r], s[1][r]), fmaxf(s[2][r], s[3][r]));
#pragma unroll
    for (int off = 1; off <= 8; off <<= 1)
#pragma unroll
      for (int r = 0; r < 8; ++r) mx[r] = fmaxf(mx[r], __shfl_xor(mx[r], off, 32));

    float al[8], rs[8];
#pragma unroll
    for (int r = 0; r < 8; ++r) {
      float mn = fmaxf(m[r], mx[r]);
      al[r] = exp2f(m[r] - mn);
      m[r] = mn;
      rs[r] = 0.0f;
    }
#pragma unroll
    for (int t = 0; t < 4; ++t)
#pragma unroll
      for (int r = 0; r < 8; ++r) {
        float p = exp2f(s[t][r] - m[r]);
        s[t][r] = p;
        rs[r] += p;
      }
#pragma unroll
    for (int off = 1; off <= 8; off <<= 1)
#pragma unroll
      for (int r = 0; r < 8; ++r) rs[r] += __shfl_xor(rs[r], off, 32);
#pragma unroll
    for (int r = 0; r < 8; ++r) l[r] = l[r] * al[r] + rs[r];
#pragma unroll
    for (int j = 0; j < 4; ++j)
#pragma unroll
      for (int r = 0; r < 8; ++r) o[j][r] *= al[r];

    // ---- P: C-layout -> A-layout via wave-private LDS bounce ----
#pragma unroll
    for (int t = 0; t < 4; ++t)
#pragma unroll
      for (int r = 0; r < 8; ++r)
        mylds[(r + (hl << 3)) * P_STRIDE + t * 16 + ln] = (_Float16)s[t][r];
    asm volatile("s_wait_dscnt 0" ::: "memory");
    const _Float16* prow = mylds + ln * P_STRIDE;
    v16h ap0 = mk16(*(const h8*)(prow + hl * 8), *(const h8*)(prow + hl * 8 + 16));
    v16h ap1 = mk16(*(const h8*)(prow + 32 + hl * 8), *(const h8*)(prow + 48 + hl * 8));

    // ---- O += P * V : K-dim = 64 keys (2 chained WMMAs) x 4 head tiles ----
#pragma unroll
    for (int j = 0; j < 4; ++j) {
      const _Float16* vrp = vtb + (size_t)(j * 16 + ln) * 4096 + key0;
      v16h bv0 = mk16(*(const h8*)(vrp + hl * 8), *(const h8*)(vrp + hl * 8 + 16));
      v16h bv1 = mk16(*(const h8*)(vrp + 32 + hl * 8), *(const h8*)(vrp + 48 + hl * 8));
      o[j] = __builtin_amdgcn_wmma_f32_16x16x32_f16(false, ap0, false, bv0,
                                                    (short)0, o[j], false, false);
      o[j] = __builtin_amdgcn_wmma_f32_16x16x32_f16(false, ap1, false, bv1,
                                                    (short)0, o[j], false, false);
    }
  }

  // ---- epilogue: normalize and store f32 output [b][4096][50] ----
  float inv[8];
#pragma unroll
  for (int r = 0; r < 8; ++r) inv[r] = 1.0f / l[r];
#pragma unroll
  for (int j = 0; j < 4; ++j) {
    int h = j * 16 + ln;
    if (h < 50) {
#pragma unroll
      for (int r = 0; r < 8; ++r) {
        int tok = q0 + r + (hl << 3);
        out[((size_t)b * 4096 + tok) * 50 + h] = o[j][r] * inv[r];
      }
    }
  }
}

// ---------------------------------------------------------------------------
extern "C" void kernel_launch(void* const* d_in, const int* in_sizes, int n_in,
                              void* d_out, int out_size, void* d_ws,
                              size_t ws_size, hipStream_t stream) {
  const float* x = (const float*)d_in[0];      // [8,4096,256] f32
  const float* w = (const float*)d_in[1];      // [3,256,50]   f32
  float* out = (float*)d_out;                  // [8,4096,50]  f32

  _Float16* qh = (_Float16*)d_ws;                       // [32768][64]
  _Float16* kh = qh + (size_t)32768 * 64;               // [32768][64]
  _Float16* vt = kh + (size_t)32768 * 64;               // [8][64][4096]

  qkv_kernel<<<dim3(256, 3), 256, 0, stream>>>(x, w, qh, kh, vt);
  attn_kernel<<<dim3(32, 8), 256, 0, stream>>>(qh, kh, vt, out);
}
